// FastSpeech2Model_76991583748690
// MI455X (gfx1250) — matrix operand order repaired and checked by reading, not compiled, not run on hardware
//
#include <hip/hip_runtime.h>

typedef __attribute__((ext_vector_type(16))) _Float16 v16h;
typedef __attribute__((ext_vector_type(8)))  _Float16 v8h;
typedef __attribute__((ext_vector_type(8)))  float    v8f;
typedef __attribute__((ext_vector_type(4)))  float    v4f;

// ---------------------------------------------------------------------------
// gfx1250 wave32 WMMA fragment layouts (CDNA5 ISA 7.12.2):
//  A 16x32 f16 : lane L (row=L&15, half=L>>4): elem i<8 -> K=i+half*8,
//                i>=8 -> K=(i-8)+16+half*8  => two contiguous 8-runs.
//  B 32x16 f16 : lane L (col=L&15): same K order as A  => two contiguous 8-runs.
//  C/D f32     : lane L: col=L&15, VGPR r -> row (L>>4)*8 + r.
// ---------------------------------------------------------------------------

__device__ inline v4f ld4(const float* p) { return *(const v4f*)p; }

__device__ inline v16h cvt16(v4f a, v4f b, v4f c, v4f d) {
    v16h r;
#pragma unroll
    for (int i = 0; i < 4; ++i) {
        r[i]      = (_Float16)a[i];
        r[4 + i]  = (_Float16)b[i];
        r[8 + i]  = (_Float16)c[i];
        r[12 + i] = (_Float16)d[i];
    }
    return r;
}

__device__ inline v16h ld16h(const _Float16* p) {
    v8h lo = *(const v8h*)p;
    v8h hi = *(const v8h*)(p + 16);
    return __builtin_shufflevector(lo, hi, 0,1,2,3,4,5,6,7,8,9,10,11,12,13,14,15);
}

// ---------------- GEMM: C[M,N] = act(A[M,K] @ W[N,K]^T + bias) --------------
// One wave -> 16 x (16*NT) output tile; A fragment reused across NT wmmas.
template <int NT>
__global__ __launch_bounds__(32) void k_gemm(const float* __restrict__ A,
                                             const float* __restrict__ W,
                                             const float* __restrict__ bias,
                                             float* __restrict__ C,
                                             int M, int N, int K, int act)
{
    const int ng   = N / (16 * NT);
    const int m0   = (blockIdx.x / ng) << 4;
    const int n0   = (blockIdx.x % ng) * (16 * NT);
    const int lane = threadIdx.x;
    const int row  = lane & 15;
    const int half = lane >> 4;

    const float* arow = A + (size_t)(m0 + row) * K;
    const float* brow[NT];
#pragma unroll
    for (int nt = 0; nt < NT; ++nt) brow[nt] = W + (size_t)(n0 + nt * 16 + row) * K;

    v8f zero = {};
    v8f acc[NT];
#pragma unroll
    for (int nt = 0; nt < NT; ++nt) acc[nt] = zero;

    for (int k0 = 0; k0 < K; k0 += 32) {
        const float* ap = arow + k0 + half * 8;
        v16h af = cvt16(ld4(ap), ld4(ap + 4), ld4(ap + 16), ld4(ap + 20));
#pragma unroll
        for (int nt = 0; nt < NT; ++nt) {
            const float* bp = brow[nt] + k0 + half * 8;
            v16h bf = cvt16(ld4(bp), ld4(bp + 4), ld4(bp + 16), ld4(bp + 20));
            acc[nt] = __builtin_amdgcn_wmma_f32_16x16x32_f16(false, af, false, bf, (short)0, acc[nt], false, false);
        }
    }
    const int rbase = half * 8;
#pragma unroll
    for (int nt = 0; nt < NT; ++nt) {
        int nc = n0 + nt * 16 + row;
        float bv = bias[nc];
#pragma unroll
        for (int r = 0; r < 8; ++r) {
            float v = acc[nt][r] + bv;
            if (act == 1) v = fmaxf(v, 0.f);
            C[(size_t)(m0 + rbase + r) * N + nc] = v;
        }
    }
}

// ---- conv weight transpose+f16: Wt[(j*Cout+o)*Cin+i] = W[o*Cin*KW+i*KW+j] ---
__global__ void k_wtrans(const float* __restrict__ W, _Float16* __restrict__ Wt,
                         int Cout, int Cin, int KW)
{
    int idx = blockIdx.x * 256 + threadIdx.x;
    int total = Cout * Cin * KW;
    if (idx >= total) return;
    int j  = idx % KW;
    int r  = idx / KW;
    int ci = r % Cin;
    int co = r / Cin;
    Wt[((size_t)j * Cout + co) * Cin + ci] = (_Float16)W[idx];
}

// ---- conv1d (NHC): Y[b,t,o] = sum_{j,i} X[b,t+j-pad,i] * Wt[j][o][i] -------
template <int NT>
__global__ __launch_bounds__(32) void k_conv1d(const float* __restrict__ X,
                                               const _Float16* __restrict__ Wt,
                                               const float* __restrict__ bias,
                                               float* __restrict__ Y,
                                               int S, int Cin, int Cout, int KW, int act)
{
    const int ng   = Cout / (16 * NT);
    const int m0   = (blockIdx.x / ng) << 4;
    const int n0   = (blockIdx.x % ng) * (16 * NT);
    const int b    = m0 / S;
    const int t0   = m0 % S;
    const int pad  = KW >> 1;
    const int lane = threadIdx.x;
    const int row  = lane & 15;
    const int half = lane >> 4;

    v8f zero = {};
    v8f acc[NT];
#pragma unroll
    for (int nt = 0; nt < NT; ++nt) acc[nt] = zero;
    v4f z4 = {};

    for (int j = 0; j < KW; ++j) {
        int t = t0 + row + j - pad;
        bool ok = (t >= 0) && (t < S);
        const float* arow = X + ((size_t)b * S + t) * Cin;
        for (int c0 = 0; c0 < Cin; c0 += 32) {
            const float* ap = arow + c0 + half * 8;
            v16h af = ok ? cvt16(ld4(ap), ld4(ap + 4), ld4(ap + 16), ld4(ap + 20))
                         : cvt16(z4, z4, z4, z4);
#pragma unroll
            for (int nt = 0; nt < NT; ++nt) {
                const _Float16* bp = Wt + ((size_t)j * Cout + n0 + nt * 16 + row) * Cin + c0 + half * 8;
                v16h bf = ld16h(bp);
                acc[nt] = __builtin_amdgcn_wmma_f32_16x16x32_f16(false, af, false, bf, (short)0, acc[nt], false, false);
            }
        }
    }
    const int rbase = half * 8;
#pragma unroll
    for (int nt = 0; nt < NT; ++nt) {
        int nc = n0 + nt * 16 + row;
        float bv = bias[nc];
#pragma unroll
        for (int r = 0; r < 8; ++r) {
            float v = acc[nt][r] + bv;
            if (act == 1) v = fmaxf(v, 0.f);
            Y[(size_t)(m0 + rbase + r) * Cout + nc] = v;
        }
    }
}

// ---- V transpose+f16: Vt[((b*NH+h)*d + dl)*S + key] = qkv[b,key,2H+h*d+dl] --
__global__ void k_vtrans(const float* __restrict__ qkv, _Float16* __restrict__ Vt,
                         int S, int H, int NH)
{
    int key = blockIdx.x, b = blockIdx.y, dd = threadIdx.x;   // H threads
    float v = qkv[((size_t)b * S + key) * 3 * H + 2 * H + dd];
    Vt[((size_t)b * H + dd) * S + key] = (_Float16)v;         // (b*NH+h)*d + dl == b*H + dd
}

// -------- fused MHA (flash style): per-wave (b, head, 16-query tile) --------
__global__ __launch_bounds__(32) void k_attn(const float* __restrict__ qkv,
                                             const _Float16* __restrict__ Vt,
                                             const int* __restrict__ vlen,
                                             float* __restrict__ out,
                                             int S, int H, int NH)
{
    __shared__ float    sS[16 * 32];
    __shared__ _Float16 sP[16 * 32];
    __shared__ float    sM[16], sL[16], sC[16];

    const int d    = H / NH;                  // 128
    const int q0   = blockIdx.x << 4;
    const int h    = blockIdx.y;
    const int b    = blockIdx.z;
    const int lane = threadIdx.x;
    const int row  = lane & 15;
    const int half = lane >> 4;
    const int H3   = 3 * H;
    const float scale = rsqrtf((float)d);

    const float* qb = qkv + (size_t)b * S * H3 + (size_t)h * d;
    const float* kb = qb + H;
    const int VL = vlen[b];

    // Q fragments: 4 chunks of K=32 over d, pre-scaled by 1/sqrt(d)
    v16h qa[4];
    {
        const float* qrow = qb + (size_t)(q0 + row) * H3;
#pragma unroll
        for (int c = 0; c < 4; ++c) {
            const float* qp = qrow + c * 32 + half * 8;
            qa[c] = cvt16(ld4(qp) * scale, ld4(qp + 4) * scale,
                          ld4(qp + 16) * scale, ld4(qp + 20) * scale);
        }
    }
    v8f zero = {};
    v8f oacc[8];
#pragma unroll
    for (int dc = 0; dc < 8; ++dc) oacc[dc] = zero;

    if (lane < 16) { sM[lane] = -1e30f; sL[lane] = 0.f; }
    __syncthreads();

    for (int k0 = 0; k0 < S; k0 += 32) {
        // scores: two 16x16 key tiles, contraction over d=128
#pragma unroll
        for (int kh = 0; kh < 2; ++kh) {
            int key = k0 + kh * 16 + row;                 // B col = key
            const float* krow = kb + (size_t)key * H3;
            v8f sacc = zero;
#pragma unroll
            for (int c = 0; c < 4; ++c) {
                const float* kp = krow + c * 32 + half * 8;
                v16h bf = cvt16(ld4(kp), ld4(kp + 4), ld4(kp + 16), ld4(kp + 20));
                sacc = __builtin_amdgcn_wmma_f32_16x16x32_f16(false, qa[c], false, bf, (short)0, sacc, false, false);
            }
            int rbase = half * 8;
#pragma unroll
            for (int r = 0; r < 8; ++r) {
                float v = (key < VL) ? sacc[r] : -1e9f;
                sS[(rbase + r) * 32 + kh * 16 + row] = v;
            }
        }
        __syncthreads();
        // online softmax, one row per lane 0..15 (vectorized LDS access)
        if (lane < 16) {
            const v4f* rp = (const v4f*)&sS[lane * 32];
            v4f rv[8];
            float mo = sM[lane], mn = mo;
#pragma unroll
            for (int g = 0; g < 8; ++g) {
                rv[g] = rp[g];
#pragma unroll
                for (int e = 0; e < 4; ++e) mn = fmaxf(mn, rv[g][e]);
            }
            float corr = __expf(mo - mn);
            float l = sL[lane] * corr;
#pragma unroll
            for (int g = 0; g < 4; ++g) {
                v8h pk;
#pragma unroll
                for (int e = 0; e < 8; ++e) {
                    float p = __expf(rv[g * 2 + (e >> 2)][e & 3] - mn);
                    l += p;
                    pk[e] = (_Float16)p;
                }
                *(v8h*)&sP[lane * 32 + g * 8] = pk;
            }
            sM[lane] = mn; sL[lane] = l; sC[lane] = corr;
        }
        __syncthreads();
        // rescale running output, then P @ V
        {
            int rbase = half * 8;
#pragma unroll
            for (int r = 0; r < 8; ++r) {
                float c = sC[rbase + r];
#pragma unroll
                for (int dc = 0; dc < 8; ++dc) oacc[dc][r] *= c;
            }
        }
        v16h pa;
        {
            v8h lo = *(const v8h*)&sP[row * 32 + half * 8];
            v8h hi = *(const v8h*)&sP[row * 32 + 16 + half * 8];
            pa = __builtin_shufflevector(lo, hi, 0,1,2,3,4,5,6,7,8,9,10,11,12,13,14,15);
        }
        const _Float16* vbase = Vt + (size_t)b * H * S + (size_t)h * d * S + k0 + half * 8;
#pragma unroll
        for (int dc = 0; dc < 8; ++dc) {
            v16h bf = ld16h(vbase + (size_t)(dc * 16 + row) * S);
            oacc[dc] = __builtin_amdgcn_wmma_f32_16x16x32_f16(false, pa, false, bf, (short)0, oacc[dc], false, false);
        }
        __syncthreads();
    }
    const int rbase = half * 8;
#pragma unroll
    for (int dc = 0; dc < 8; ++dc)
#pragma unroll
        for (int r = 0; r < 8; ++r) {
            float l = sL[rbase + r];
            out[((size_t)b * S + q0 + rbase + r) * H + h * d + dc * 16 + row] = oacc[dc][r] / l;
        }
}

// ---------------- LayerNorm: Y = LN(X (+R)) * g + b ; wave per row ----------
__global__ __launch_bounds__(32) void k_layernorm(const float* __restrict__ X,
                                                  const float* __restrict__ R,
                                                  const float* __restrict__ g,
                                                  const float* __restrict__ bt,
                                                  float* __restrict__ Y, int H)
{
    const size_t rowoff = (size_t)blockIdx.x * H;
    const int lane = threadIdx.x;
    float vals[8];
    float s = 0.f, ss = 0.f;
    const int n = H >> 5;
    for (int i = 0; i < n; ++i) {
        int c = lane + i * 32;
        float v = X[rowoff + c] + (R ? R[rowoff + c] : 0.f);
        vals[i] = v; s += v; ss += v * v;
    }
    for (int off = 16; off; off >>= 1) {
        s  += __shfl_xor(s,  off, 32);
        ss += __shfl_xor(ss, off, 32);
    }
    float mean = s / (float)H;
    float var  = ss / (float)H - mean * mean;
    float inv  = rsqrtf(var + 1e-5f);
    for (int i = 0; i < n; ++i) {
        int c = lane + i * 32;
        Y[rowoff + c] = (vals[i] - mean) * inv * g[c] + bt[c];
    }
}

// ---------------- embedding + sinusoidal PE ---------------------------------
__global__ void k_embed(const int* __restrict__ ids, const float* __restrict__ emb,
                        float* __restrict__ X, int T, int H)
{
    int bt = blockIdx.x, h = threadIdx.x;
    int t = bt % T;
    float arg = (float)t * __expf(-(float)(h & ~1) * (9.210340372f / (float)H));
    float pe = (h & 1) ? __cosf(arg) : __sinf(arg);
    X[(size_t)bt * H + h] = emb[(size_t)ids[bt] * H + h] + pe;
}

// ---------------- pitch/energy bucketize + embedding add --------------------
__global__ void k_addemb(const float* __restrict__ enc, const float* __restrict__ pitch,
                         const float* __restrict__ energy, const float* __restrict__ pemb,
                         const float* __restrict__ eemb, float* __restrict__ out, int H)
{
    int bt = blockIdx.x, h = threadIdx.x;
    float pv = fminf(fmaxf(pitch[bt],  -3.f), 3.f);
    float ev = fminf(fmaxf(energy[bt], -3.f), 3.f);
    int pi = (int)ceilf((pv + 3.f) * (255.f / 6.f));
    int ei = (int)ceilf((ev + 3.f) * (255.f / 6.f));
    pi = pi < 0 ? 0 : (pi > 255 ? 255 : pi);
    ei = ei < 0 ? 0 : (ei > 255 ? 255 : ei);
    out[(size_t)bt * H + h] = enc[(size_t)bt * H + h] + pemb[(size_t)pi * H + h] + eemb[(size_t)ei * H + h];
}

// ---------------- duration cumsum + mel_len ---------------------------------
__global__ void k_cumsum(const int* __restrict__ dur, int* __restrict__ cum,
                         int* __restrict__ mlen, float* __restrict__ mlen_f, int T)
{
    int b = blockIdx.x;
    if (threadIdx.x == 0) {
        int s = 0;
        for (int t = 0; t < T; ++t) { s += dur[b * T + t]; cum[b * T + t] = s; }
        mlen[b] = s;
        mlen_f[b] = (float)s;
    }
}

// ---------------- length regulator expansion + PE ---------------------------
__global__ void k_expand(const float* __restrict__ xad, const int* __restrict__ cum,
                         const int* __restrict__ mlen, float* __restrict__ xexp,
                         int T, int H, int MMEL)
{
    int pos = blockIdx.x, b = blockIdx.y, h = threadIdx.x;
    const int* c = cum + b * T;
    int lo = 0, hi = T;
    while (lo < hi) { int mid = (lo + hi) >> 1; if (c[mid] <= pos) lo = mid + 1; else hi = mid; }
    int idx = lo < (T - 1) ? lo : (T - 1);
    float v = (pos < mlen[b]) ? xad[((size_t)b * T + idx) * H + h] : 0.f;
    float arg = (float)pos * __expf(-(float)(h & ~1) * (9.210340372f / (float)H));
    v += (h & 1) ? __cosf(arg) : __sinf(arg);
    xexp[((size_t)b * MMEL + pos) * H + h] = v;
}

// ---------------- scalar projection (N=1) -----------------------------------
__global__ __launch_bounds__(32) void k_proj1(const float* __restrict__ X,
                                              const float* __restrict__ w,
                                              const float* __restrict__ b0,
                                              float* __restrict__ out, int H)
{
    size_t rowoff = (size_t)blockIdx.x * H;
    int lane = threadIdx.x;
    float s = 0.f;
    for (int i = lane; i < H; i += 32) s += X[rowoff + i] * w[i];
    for (int off = 16; off; off >>= 1) s += __shfl_xor(s, off, 32);
    if (lane == 0) out[blockIdx.x] = s + b0[0];
}

// ===========================================================================
extern "C" void kernel_launch(void* const* d_in, const int* in_sizes, int n_in,
                              void* d_out, int out_size, void* d_ws, size_t ws_size,
                              hipStream_t stream)
{
    (void)in_sizes; (void)n_in; (void)out_size; (void)ws_size;
    const int B = 16, T = 256, H = 256, FF = 1024, MMEL = 1024, NH = 2, MEL = 80;

    auto F = [&](int i) { return (const float*)d_in[i]; };
    const int*   ids    = (const int*)d_in[0];
    const int*   lens   = (const int*)d_in[1];
    const int*   durs   = (const int*)d_in[2];
    const float* pitch  = F(3);
    const float* energy = F(4);
    const float* embed  = F(5);

    float* ws = (float*)d_ws;
    const size_t Rmax = (size_t)B * MMEL;                 // 16384 rows max
    float*     bufX   = ws;                               // [Rmax, H]
    float*     bufA   = bufX   + Rmax * H;                // [Rmax, H]
    float*     bufT   = bufA   + Rmax * H;                // [Rmax, H]
    float*     bufQKV = bufT   + Rmax * H;                // [Rmax, 3H]
    float*     bufFF  = bufQKV + Rmax * 3 * H;            // [Rmax, FF]
    float*     bufV1  = bufFF  + Rmax * FF;               // [B*T, H]
    float*     bufV2  = bufV1  + (size_t)B * T * H;       // [B*T, H]
    _Float16*  Wt     = (_Float16*)(bufV2 + (size_t)B * T * H);   // [9*FF*H] f16
    _Float16*  Vt     = Wt + (size_t)9 * FF * H;          // [B*H*MMEL] f16
    int*       cum    = (int*)(Vt + Rmax * H);
    int*       mlen   = cum + B * T;

    float* outMel    = (float*)d_out;
    float* outDur    = outMel + (size_t)B * MMEL * MEL;
    float* outPitch  = outDur + B * T;
    float* outEnergy = outPitch + B * T;
    float* outMlen   = outEnergy + B * T;

    // embedding + positional encoding -> bufX [B*T, H]
    k_embed<<<B * T, H, 0, stream>>>(ids, embed, bufX, T, H);

    auto fft_block = [&](float* X, int S, const int* vlen, int pb) {
        int M = B * S;
        k_gemm<4><<<(M / 16) * (3 * H / 64), 32, 0, stream>>>(X, F(pb + 0), F(pb + 1), bufQKV, M, 3 * H, H, 0);
        k_vtrans<<<dim3(S, B), H, 0, stream>>>(bufQKV, Vt, S, H, NH);
        k_attn<<<dim3(S / 16, NH, B), 32, 0, stream>>>(bufQKV, Vt, vlen, bufA, S, H, NH);
        k_gemm<4><<<(M / 16) * (H / 64), 32, 0, stream>>>(bufA, F(pb + 2), F(pb + 3), bufT, M, H, H, 0);
        k_layernorm<<<M, 32, 0, stream>>>(X, bufT, F(pb + 4), F(pb + 5), X, H);
        k_wtrans<<<(FF * H * 9 + 255) / 256, 256, 0, stream>>>(F(pb + 6), Wt, FF, H, 9);
        k_conv1d<4><<<(M / 16) * (FF / 64), 32, 0, stream>>>(X, Wt, F(pb + 7), bufFF, S, H, FF, 9, 1);
        k_wtrans<<<(H * FF * 9 + 255) / 256, 256, 0, stream>>>(F(pb + 8), Wt, H, FF, 9);
        k_conv1d<4><<<(M / 16) * (H / 64), 32, 0, stream>>>(bufFF, Wt, F(pb + 9), bufT, S, FF, H, 9, 0);
        k_layernorm<<<M, 32, 0, stream>>>(X, bufT, F(pb + 10), F(pb + 11), X, H);
    };

    // encoder: param indices 6 + i*12
    for (int i = 0; i < 4; ++i) fft_block(bufX, T, lens, 6 + i * 12);

    // variance predictors (read enc_out in bufX)
    auto var_pred = [&](int pb, float* outv) {
        int M = B * T;
        k_wtrans<<<(H * H * 3 + 255) / 256, 256, 0, stream>>>(F(pb + 0), Wt, H, H, 3);
        k_conv1d<4><<<(M / 16) * (H / 64), 32, 0, stream>>>(bufX, Wt, F(pb + 1), bufV1, T, H, H, 3, 1);
        k_layernorm<<<M, 32, 0, stream>>>(bufV1, nullptr, F(pb + 4), F(pb + 5), bufV1, H);
        k_wtrans<<<(H * H * 3 + 255) / 256, 256, 0, stream>>>(F(pb + 2), Wt, H, H, 3);
        k_conv1d<4><<<(M / 16) * (H / 64), 32, 0, stream>>>(bufV1, Wt, F(pb + 3), bufV2, T, H, H, 3, 1);
        k_layernorm<<<M, 32, 0, stream>>>(bufV2, nullptr, F(pb + 6), F(pb + 7), bufV2, H);
        k_proj1<<<M, 32, 0, stream>>>(bufV2, F(pb + 8), F(pb + 9), outv, H);
    };
    var_pred(102, outDur);
    var_pred(112, outPitch);
    var_pred(122, outEnergy);

    // x_ad = enc_out + pitch_emb[idx] + energy_emb[idx]  -> bufA
    k_addemb<<<B * T, H, 0, stream>>>(bufX, pitch, energy, F(132), F(133), bufA, H);

    // length regulator: cumsum + expand -> bufX [B*MMEL, H]
    k_cumsum<<<B, 1, 0, stream>>>(durs, cum, mlen, outMlen, T);
    k_expand<<<dim3(MMEL, B), H, 0, stream>>>(bufA, cum, mlen, bufX, T, H, MMEL);

    // decoder: param indices 54 + i*12
    for (int i = 0; i < 4; ++i) fft_block(bufX, MMEL, mlen, 54 + i * 12);

    // mel projection: [B*MMEL, 80] with NT=5 (80 = 5*16)
    k_gemm<5><<<((B * MMEL) / 16) * 1, 32, 0, stream>>>(bufX, F(134), F(135), outMel, B * MMEL, MEL, H, 0);
}
